// Net_89343909691621
// MI455X (gfx1250) — matrix operand order repaired and checked by reading, not compiled, not run on hardware
//
#include <hip/hip_runtime.h>
#include <cstdint>
#include <cstddef>

#define BB 16384
#define DD 2048
#define HH 256
#define OO 1000
#define EE 8
#define MT 64                 // rows (M) per workgroup
#define NTHR 512              // 16 waves per workgroup

typedef __attribute__((ext_vector_type(16))) __bf16 v16bf;
typedef __attribute__((ext_vector_type(8)))  __bf16 v8bf;
typedef __attribute__((ext_vector_type(8)))  float  v8f;

constexpr int XCH = 40;       // x K-chunk LDS row stride (halfs): 32 + 8 pad
constexpr int HST = HH + 8;   // h LDS row stride (halfs)

// A-fragment (16x32 bf16): lane m=l&15, 16 consecutive k [kb..kb+16)
__device__ __forceinline__ v16bf load_a_lds(const __bf16* p) {
    v8bf lo = *(const v8bf*)p;
    v8bf hi = *(const v8bf*)(p + 8);
    return __builtin_shufflevector(lo, hi, 0,1,2,3,4,5,6,7,8,9,10,11,12,13,14,15);
}
// B-fragment: 16 consecutive bf16 K-values of one weight row (32B aligned)
__device__ __forceinline__ v16bf load_b_bf16(const __bf16* p) {
    return *(const v16bf*)p;
}

// ---------------- one-time f32 -> bf16 weight conversion --------------------
__global__ __launch_bounds__(256) void cvt_bf16_kernel(
    const float* __restrict__ src, __bf16* __restrict__ dst, int n)
{
    const int i = (blockIdx.x * 256 + threadIdx.x) * 8;
    if (i < n) {
        float4 a = *(const float4*)(src + i);
        float4 b = *(const float4*)(src + i + 4);
        v8bf o;
        o[0] = (__bf16)a.x; o[1] = (__bf16)a.y; o[2] = (__bf16)a.z; o[3] = (__bf16)a.w;
        o[4] = (__bf16)b.x; o[5] = (__bf16)b.y; o[6] = (__bf16)b.z; o[7] = (__bf16)b.w;
        *(v8bf*)(dst + i) = o;
    }
}

// ---------------- Gating: logits -> softmax -> top2 mask -> L1 normalize ----
__global__ __launch_bounds__(256) void gate_kernel(
    const float* __restrict__ x, const float* __restrict__ gw,
    const float* __restrict__ gb, float* __restrict__ gs)
{
    const int row  = blockIdx.x * 8 + (threadIdx.x >> 5);
    const int lane = threadIdx.x & 31;
    const float* xr = x + (size_t)row * DD;

    float acc[EE];
#pragma unroll
    for (int e = 0; e < EE; ++e) acc[e] = 0.f;

    for (int k = lane * 4; k < DD; k += 128) {
        float4 xv = *(const float4*)(xr + k);
#pragma unroll
        for (int e = 0; e < EE; ++e) {
            float4 wv = *(const float4*)(gw + (size_t)e * DD + k);
            acc[e] += xv.x * wv.x + xv.y * wv.y + xv.z * wv.z + xv.w * wv.w;
        }
    }
#pragma unroll
    for (int e = 0; e < EE; ++e)
        for (int off = 16; off >= 1; off >>= 1)
            acc[e] += __shfl_xor(acc[e], off, 32);

    float mx = -1e30f;
#pragma unroll
    for (int e = 0; e < EE; ++e) { acc[e] += gb[e]; mx = fmaxf(mx, acc[e]); }
    float g[EE], s = 0.f;
#pragma unroll
    for (int e = 0; e < EE; ++e) { g[e] = __expf(acc[e] - mx); s += g[e]; }
    const float invs = 1.f / s;
#pragma unroll
    for (int e = 0; e < EE; ++e) g[e] *= invs;

    float m1 = -1.f, m2 = -1.f;          // top-2 values
#pragma unroll
    for (int e = 0; e < EE; ++e) {
        if (g[e] > m1)      { m2 = m1; m1 = g[e]; }
        else if (g[e] > m2) { m2 = g[e]; }
    }
    float l1 = 0.f;
#pragma unroll
    for (int e = 0; e < EE; ++e) { if (g[e] < m2) g[e] = 0.f; l1 += g[e]; }
    const float invl1 = 1.f / fmaxf(l1, 1e-12f);
    if (lane < EE) gs[(size_t)row * EE + lane] = g[lane] * invl1;
}

// ---------------- Fused dual-GEMM MoE (64 rows / WG, 16 waves) --------------
__global__ __launch_bounds__(NTHR) void moe_kernel(
    const float*  __restrict__ x,
    const __bf16* __restrict__ w1b, const float* __restrict__ b1,
    const __bf16* __restrict__ w2b, const float* __restrict__ b2,
    const float*  __restrict__ gs,  float* __restrict__ out)
{
    __shared__ alignas(16) __bf16 xs[2][MT * XCH];  // double-buffered x K-chunk
    __shared__ alignas(16) __bf16 hs[MT * HST];     // gated+relu'd h tile
    __shared__ float gss[MT * EE];

    const int tid  = threadIdx.x;
    const int wave = tid >> 5;                      // 0..15
    const int lane = tid & 31;
    const int l16  = lane & 15;
    const int lhi  = lane >> 4;
    const int row0 = blockIdx.x * MT;

    gss[tid] = gs[(size_t)row0 * EE + tid];         // 512 == MT*EE

    // x-staging coords: 64x32 f32 chunk, one float4 per thread
    const int sr = tid >> 3;                        // row 0..63
    const int sc = (tid & 7) << 2;                  // col 0,4,..,28
    const float* xrow = x + (size_t)(row0 + sr) * DD + sc;

    v8f oacc[4][4] = {};                            // 64 rows x 64 cols per wave

    for (int e = 0; e < EE; ++e) {
        // ---- stage 1: h[64,256] = relu(x @ w1[e]^T + b1) * gs --------------
        const __bf16* w1e = w1b + (size_t)e * HH * DD;
        const int n0 = wave * 16 + l16;             // wave owns 16 h-columns
        v8f hacc[4] = {};

        {   // prologue: stage chunk k0=0 into buffer 0
            float4 v = *(const float4*)xrow;
            __bf16* p = &xs[0][sr * XCH + sc];
            p[0] = (__bf16)v.x; p[1] = (__bf16)v.y;
            p[2] = (__bf16)v.z; p[3] = (__bf16)v.w;
        }
        __syncthreads();

        int buf = 0;
        for (int k0 = 0; k0 < DD; k0 += 32) {
            if (k0 + 32 < DD) {                     // stage next chunk
                float4 v = *(const float4*)(xrow + k0 + 32);
                __bf16* p = &xs[buf ^ 1][sr * XCH + sc];
                p[0] = (__bf16)v.x; p[1] = (__bf16)v.y;
                p[2] = (__bf16)v.z; p[3] = (__bf16)v.w;
            }
            v16bf bA = load_b_bf16(w1e + (size_t)n0 * DD + k0 + lhi * 16);
#pragma unroll
            for (int mb = 0; mb < 4; ++mb) {
                v16bf a = load_a_lds(&xs[buf][(mb * 16 + l16) * XCH + lhi * 16]);
                hacc[mb] = __builtin_amdgcn_wmma_f32_16x16x32_bf16(
                    false, a, false, bA, (short)0, hacc[mb], false, false);
            }
            if (k0 + 64 < DD)
                __builtin_prefetch(w1e + (size_t)n0 * DD + k0 + 64 + lhi * 16, 0, 1);
            __syncthreads();
            buf ^= 1;
        }

        const float bias0 = b1[e * HH + n0];
#pragma unroll
        for (int mb = 0; mb < 4; ++mb)
#pragma unroll
            for (int r = 0; r < 8; ++r) {
                const int m  = mb * 16 + r + 8 * lhi;   // C/D layout
                const float g = gss[m * EE + e];
                float v0 = hacc[mb][r] + bias0;
                v0 = v0 > 0.f ? v0 * g : 0.f;
                hs[m * HST + n0] = (__bf16)v0;
            }
        __syncthreads();                            // h visible to all waves

        // ---- stage 2: out += (gs*h) @ w2[e]^T; wave owns cols [wave*64,+64)
        const __bf16* w2e = w2b + (size_t)e * OO * HH;
        for (int k0 = 0; k0 < HH; k0 += 32) {
            v16bf a[4];
#pragma unroll
            for (int mb = 0; mb < 4; ++mb)
                a[mb] = load_a_lds(&hs[(mb * 16 + l16) * HST + k0 + lhi * 16]);
#pragma unroll
            for (int t = 0; t < 4; ++t) {
                const int n  = wave * 64 + t * 16 + l16;
                const int nc = n < OO ? n : (OO - 1);   // pad cols, masked on store
                v16bf b = load_b_bf16(w2e + (size_t)nc * HH + k0 + lhi * 16);
#pragma unroll
                for (int mb = 0; mb < 4; ++mb)
                    oacc[mb][t] = __builtin_amdgcn_wmma_f32_16x16x32_bf16(
                        false, a[mb], false, b, (short)0, oacc[mb][t], false, false);
            }
        }
        __syncthreads();                            // before hs is overwritten
    }

    // ---- epilogue: out += sum_e gs[m,e]*b2[e,n]; store f32, mask N=1000 ----
#pragma unroll
    for (int t = 0; t < 4; ++t) {
        const int n = wave * 64 + t * 16 + l16;
        if (n < OO) {
            float b2v[EE];
#pragma unroll
            for (int e = 0; e < EE; ++e) b2v[e] = b2[e * OO + n];
#pragma unroll
            for (int mb = 0; mb < 4; ++mb)
#pragma unroll
                for (int r = 0; r < 8; ++r) {
                    const int m = mb * 16 + r + 8 * lhi;
                    float bt = 0.f;
#pragma unroll
                    for (int e = 0; e < EE; ++e) bt += gss[m * EE + e] * b2v[e];
                    out[(size_t)(row0 + m) * OO + n] = oacc[mb][t][r] + bt;
                }
        }
    }
}

extern "C" void kernel_launch(void* const* d_in, const int* in_sizes, int n_in,
                              void* d_out, int out_size, void* d_ws, size_t ws_size,
                              hipStream_t stream) {
    const float* x  = (const float*)d_in[0];
    const float* gw = (const float*)d_in[1];
    const float* gb = (const float*)d_in[2];
    const float* w1 = (const float*)d_in[3];
    const float* b1 = (const float*)d_in[4];
    const float* w2 = (const float*)d_in[5];
    const float* b2 = (const float*)d_in[6];
    float* out = (float*)d_out;

    // workspace layout: gs (512 KB) | w1 bf16 (8 MB) | w2 bf16 (4 MB)  ~13 MB
    char* ws = (char*)d_ws;
    float*  gsb = (float*)ws;
    __bf16* w1b = (__bf16*)(ws + (size_t)BB * EE * 4);
    __bf16* w2b = (__bf16*)(ws + (size_t)BB * EE * 4 + (size_t)EE * HH * DD * 2);

    const int n1 = EE * HH * DD;    // 4,194,304
    const int n2 = EE * OO * HH;    // 2,048,000
    cvt_bf16_kernel<<<n1 / 2048, 256, 0, stream>>>(w1, w1b, n1);
    cvt_bf16_kernel<<<n2 / 2048, 256, 0, stream>>>(w2, w2b, n2);
    gate_kernel<<<BB / 8, 256, 0, stream>>>(x, gw, gb, gsb);
    moe_kernel <<<BB / MT, NTHR, 0, stream>>>(x, w1b, b1, w2b, b2, gsb, out);
}
